// StoryGRU_32435593019732
// MI455X (gfx1250) — matrix operand order repaired and checked by reading, not compiled, not run on hardware
//
#include <hip/hip_runtime.h>
#include <hip/hip_bf16.h>
#include <math.h>

typedef __bf16 bf16;
typedef __attribute__((ext_vector_type(16))) __bf16 v16bf;
typedef __attribute__((ext_vector_type(8)))  float  v8f;

#define DEVI __device__ __forceinline__

static constexpr int B = 16, S = 256, E = 512, H = 1024, V = 32000;

DEVI float sigmoidf_(float x) { return 1.0f / (1.0f + __expf(-x)); }

// ---- fragment loaders -------------------------------------------------------
// A fragment: 16x32 bf16, contiguous k = 16*half + e. lane: M = lane&15, half = lane>>4.
DEVI v16bf load_afrag(const bf16* __restrict__ Ab, int lda, int mrow, int half, int kk)
{
    v16bf a;
#pragma unroll
    for (int g = 0; g < 8; ++g) {
        int kb = kk + half * 16 + g * 2;
        const bf16* p = Ab + (size_t)mrow * lda + kb;
        a[2 * g]     = p[0];
        a[2 * g + 1] = p[1];
    }
    return a;
}
// B fragment: lane holds 16 contiguous k of one weight row (row = tile N + lane&15).
DEVI v16bf load_bfrag(const float* __restrict__ q)
{
    v16bf b;
#pragma unroll
    for (int e = 0; e < 16; ++e) b[e] = (bf16)q[e];
    return b;
}
DEVI v16bf load_bfrag(const bf16* __restrict__ q)
{
    return *(const v16bf*)q;   // 32B, aligned (k offsets multiples of 16 elems)
}

// ---------------------------------------------------------------------------
// Generic gate GEMM:  C[Mtiles*16, Ncols] = [A0 | A1] @ [Wx | Wh]^T (+ bias)
// Gate block g = column/1024 selects Wx_g / Wh_g / bias_g (bias may be null).
// One M-tile per wave, NT N-tiles per wave sharing the A fragment.
// Weights: WT = bf16 (pre-converted) or float (inline cvt fallback).
// ---------------------------------------------------------------------------
template<typename WT, int NT>
__global__ void __launch_bounds__(256)
gemm_gates(const bf16* __restrict__ A0, int K0,
           const bf16* __restrict__ A1, int K1,
           const WT* __restrict__ Wx0, const WT* __restrict__ Wx1, const WT* __restrict__ Wx2,
           const WT* __restrict__ Wh0, const WT* __restrict__ Wh1, const WT* __restrict__ Wh2,
           const float* __restrict__ b0, const float* __restrict__ b1, const float* __restrict__ b2,
           float* __restrict__ C, int Ncols)
{
    const int lane = threadIdx.x & 31;
    const int wid  = (blockIdx.x * blockDim.x + threadIdx.x) >> 5;
    const int tilesPerRow = Ncols / (16 * NT);
    const int mt   = wid / tilesPerRow;
    const int tb   = (wid % tilesPerRow) * NT;
    const int mrow = lane & 15;
    const int half = lane >> 4;
    const int Ktot = K0 + K1;

    const WT* Wx[NT]; const WT* Wh[NT]; const float* bia[NT];
    int nloc[NT]; int ncol[NT];
#pragma unroll
    for (int j = 0; j < NT; ++j) {
        int nc  = (tb + j) * 16;
        ncol[j] = nc;
        int g   = nc >> 10;            // gate block of 1024 columns
        nloc[j] = nc & 1023;
        Wx[j]   = (g == 0) ? Wx0 : (g == 1) ? Wx1 : Wx2;
        Wh[j]   = (g == 0) ? Wh0 : (g == 1) ? Wh1 : Wh2;
        bia[j]  = (g == 0) ? b0  : (g == 1) ? b1  : b2;
    }

    v8f acc[NT] = {};

    for (int k0 = 0; k0 < Ktot; k0 += 32) {
        const bf16* Ab; int kk, lda;
        if (k0 < K0) { Ab = A0; kk = k0;      lda = K0; }
        else         { Ab = A1; kk = k0 - K0; lda = K1; }
        v16bf a = load_afrag(Ab, lda, mt * 16 + mrow, half, kk);

#pragma unroll
        for (int j = 0; j < NT; ++j) {
            const WT* Wsel; int kw, ldw;
            if (k0 < K0) { Wsel = Wx[j]; kw = k0;      ldw = K0; }
            else         { Wsel = Wh[j]; kw = k0 - K0; ldw = K1; }
            v16bf b = load_bfrag(Wsel + (size_t)(nloc[j] + mrow) * ldw + kw + half * 16);
            acc[j] = __builtin_amdgcn_wmma_f32_16x16x32_bf16(
                false, a, false, b, (short)0, acc[j], false, false);
        }
    }

#pragma unroll
    for (int j = 0; j < NT; ++j) {
        float bv  = bia[j] ? bia[j][nloc[j] + mrow] : 0.0f;
        int   col = ncol[j] + mrow;
#pragma unroll
        for (int v = 0; v < 8; ++v) {
            int m = mt * 16 + v + 8 * half;
            C[(size_t)m * Ncols + col] = acc[j][v] + bv;
        }
    }
}

// ---------------------------------------------------------------------------
// LM head: out[b,t,:] = outs[t*16+b,:] @ fcw^T + fcb.  NT=8 N-tiles/wave.
// ---------------------------------------------------------------------------
template<typename WT>
__global__ void __launch_bounds__(256)
fc_gemm(const bf16* __restrict__ A, const WT* __restrict__ W,
        const float* __restrict__ bias, float* __restrict__ out)
{
    constexpr int NT = 8;
    constexpr int K  = H;
    const int lane = threadIdx.x & 31;
    const int wid  = (blockIdx.x * blockDim.x + threadIdx.x) >> 5;
    const int mt   = wid / (2000 / NT);
    const int n0   = (wid % (2000 / NT)) * (16 * NT);
    const int mrow = lane & 15;
    const int half = lane >> 4;

    v8f acc[NT] = {};

    const bf16* arow = A + (size_t)(mt * 16 + mrow) * K;
    for (int k0 = 0; k0 < K; k0 += 32) {
        v16bf a = load_afrag(arow, 0 /*row folded*/, 0, half, k0); // lda unused: mrow=0
#pragma unroll
        for (int j = 0; j < NT; ++j) {
            const WT* q = W + (size_t)(n0 + j * 16 + mrow) * K + k0 + half * 16;
            __builtin_prefetch((const void*)(q + 32), 0, 1);   // next K-slab -> global_prefetch
            v16bf b = load_bfrag(q);
            acc[j] = __builtin_amdgcn_wmma_f32_16x16x32_bf16(
                false, a, false, b, (short)0, acc[j], false, false);
        }
    }

#pragma unroll
    for (int j = 0; j < NT; ++j) {
        int   col = n0 + j * 16 + mrow;
        float bv  = bias[col];
#pragma unroll
        for (int v = 0; v < 8; ++v) {
            int m  = mt * 16 + v + 8 * half;     // row = t*16 + b
            int bb = m & 15, tt = m >> 4;
            out[(size_t)(bb * S + tt) * V + col] = acc[j][v] + bv;
        }
    }
}

// --------------------------- elementwise kernels ----------------------------
__global__ void cvt_w(const float* __restrict__ src, bf16* __restrict__ dst, int n)
{
    int i = blockIdx.x * blockDim.x + threadIdx.x;
    if (i < n) dst[i] = (bf16)src[i];
}

// All-timestep embedding gather: xall row r = t*16 + b.
__global__ void xgather_all(const int* __restrict__ x, const float* __restrict__ emb,
                            bf16* __restrict__ xall)
{
    int i = blockIdx.x * blockDim.x + threadIdx.x;           // over 4096*512
    if (i >= B * S * E) return;
    int r = i >> 9, e = i & (E - 1);
    int tt = r >> 4, bb = r & 15;
    int tok = x[bb * S + tt];
    xall[i] = (bf16)emb[(size_t)tok * E + e];
}

// Per-step embedding gather (fallback path).
__global__ void xgather(const int* __restrict__ x, const float* __restrict__ emb,
                        int t, bf16* __restrict__ xbf)
{
    int i = blockIdx.x * blockDim.x + threadIdx.x;
    if (i >= B * E) return;
    int b = i >> 9, e = i & (E - 1);
    int tok = x[b * S + t];
    xbf[i] = (bf16)emb[(size_t)tok * E + e];
}

// z/r activation. x0slice: optional precomputed x-part ([16,3072], z @0, r @1024).
__global__ void gru_zr_ew(const float* __restrict__ gzr, const float* __restrict__ x0slice,
                          const float* __restrict__ h,
                          float* __restrict__ z, bf16* __restrict__ rh)
{
    int i = blockIdx.x * blockDim.x + threadIdx.x;
    if (i >= B * H) return;
    int m = i >> 10, j = i & (H - 1);
    float xz = x0slice ? x0slice[m * 3072 + j]        : 0.0f;
    float xr = x0slice ? x0slice[m * 3072 + 1024 + j] : 0.0f;
    float zv = sigmoidf_(gzr[m * 2048 + j] + xz);
    float rv = sigmoidf_(gzr[m * 2048 + 1024 + j] + xr);
    z[i]  = zv;
    rh[i] = (bf16)(rv * h[i]);
}

// h' = (1-z)*tanh(n) + z*h.  x0slice: optional n-gate x-part at offset 2048.
__global__ void gru_comb(const float* __restrict__ gn, const float* __restrict__ x0slice,
                         const float* __restrict__ z,
                         float* __restrict__ h, bf16* __restrict__ hbf,
                         bf16* __restrict__ outsRow)
{
    int i = blockIdx.x * blockDim.x + threadIdx.x;
    if (i >= B * H) return;
    int m = i >> 10, j = i & (H - 1);
    float xn = x0slice ? x0slice[m * 3072 + 2048 + j] : 0.0f;
    float nv = tanhf(gn[i] + xn);
    float zv = z[i];
    float hn = (1.0f - zv) * nv + zv * h[i];
    h[i]   = hn;
    hbf[i] = (bf16)hn;
    if (outsRow) outsRow[i] = (bf16)hn;
}

__global__ void init_state(float* h0f, float* h1f, bf16* h0b, bf16* h1b)
{
    int i = blockIdx.x * blockDim.x + threadIdx.x;
    if (i >= B * H) return;
    h0f[i] = 0.0f; h1f[i] = 0.0f;
    h0b[i] = (bf16)0.0f; h1b[i] = (bf16)0.0f;
}

__global__ void copy_states(const float* h0f, const float* h1f, float* d0, float* d1)
{
    int i = blockIdx.x * blockDim.x + threadIdx.x;
    if (i >= B * H) return;
    d0[i] = h0f[i];
    d1[i] = h1f[i];
}

// ---------------------------------------------------------------------------
extern "C" void kernel_launch(void* const* d_in, const int* in_sizes, int n_in,
                              void* d_out, int out_size, void* d_ws, size_t ws_size,
                              hipStream_t stream)
{
    const int*   x    = (const int*)  d_in[0];
    const float* emb  = (const float*)d_in[1];
    const float* wxz0 = (const float*)d_in[2],  *bxz0 = (const float*)d_in[3];
    const float* whz0 = (const float*)d_in[4];
    const float* wxr0 = (const float*)d_in[5],  *bxr0 = (const float*)d_in[6];
    const float* whr0 = (const float*)d_in[7];
    const float* wxn0 = (const float*)d_in[8],  *bxn0 = (const float*)d_in[9];
    const float* whn0 = (const float*)d_in[10];
    const float* wxz1 = (const float*)d_in[11], *bxz1 = (const float*)d_in[12];
    const float* whz1 = (const float*)d_in[13];
    const float* wxr1 = (const float*)d_in[14], *bxr1 = (const float*)d_in[15];
    const float* whr1 = (const float*)d_in[16];
    const float* wxn1 = (const float*)d_in[17], *bxn1 = (const float*)d_in[18];
    const float* whn1 = (const float*)d_in[19];
    const float* fcw  = (const float*)d_in[20], *fcb  = (const float*)d_in[21];
    (void)in_sizes; (void)n_in; (void)out_size;

    float* out    = (float*)d_out;
    float* out_h0 = out + (size_t)B * S * V;
    float* out_h1 = out_h0 + B * H;

    // ---- workspace carve-up (256B aligned). Common block first. ----
    char* ws = (char*)d_ws;
    size_t off = 0;
    auto carve = [&](size_t bytes) { void* p = ws + off; off += (bytes + 255) & ~(size_t)255; return p; };
    bf16*  xbf  = (bf16*) carve((size_t)B * E * 2);          // fallback per-step x
    float* h0f  = (float*)carve((size_t)B * H * 4);
    bf16*  h0b  = (bf16*) carve((size_t)B * H * 2);
    float* h1f  = (float*)carve((size_t)B * H * 4);
    bf16*  h1b  = (bf16*) carve((size_t)B * H * 2);
    float* z0   = (float*)carve((size_t)B * H * 4);
    float* z1   = (float*)carve((size_t)B * H * 4);
    bf16*  rh0  = (bf16*) carve((size_t)B * H * 2);
    bf16*  rh1  = (bf16*) carve((size_t)B * H * 2);
    float* gzr0 = (float*)carve((size_t)B * 2 * H * 4);
    float* gzr1 = (float*)carve((size_t)B * 2 * H * 4);
    float* gn0  = (float*)carve((size_t)B * H * 4);
    float* gn1  = (float*)carve((size_t)B * H * 4);
    bf16*  outs = (bf16*) carve((size_t)B * S * H * 2);      // rows = t*16 + b

    // Fast-path extras: all-step x (bf16), X0 (f32), bf16 weights.
    bf16*  xall = (bf16*) carve((size_t)B * S * E * 2);      // 4 MB
    float* X0   = (float*)carve((size_t)B * S * 3 * H * 4);  // 50.3 MB
    bf16*  wz0b = (bf16*) carve((size_t)H * E * 2);
    bf16*  wr0b = (bf16*) carve((size_t)H * E * 2);
    bf16*  wn0b = (bf16*) carve((size_t)H * E * 2);
    bf16*  uz0b = (bf16*) carve((size_t)H * H * 2);
    bf16*  ur0b = (bf16*) carve((size_t)H * H * 2);
    bf16*  un0b = (bf16*) carve((size_t)H * H * 2);
    bf16*  wz1b = (bf16*) carve((size_t)H * H * 2);
    bf16*  wr1b = (bf16*) carve((size_t)H * H * 2);
    bf16*  wn1b = (bf16*) carve((size_t)H * H * 2);
    bf16*  uz1b = (bf16*) carve((size_t)H * H * 2);
    bf16*  ur1b = (bf16*) carve((size_t)H * H * 2);
    bf16*  un1b = (bf16*) carve((size_t)H * H * 2);
    bf16*  fcwb = (bf16*) carve((size_t)V * H * 2);          // 65.5 MB
    const bool fast = (off <= ws_size);                       // deterministic per launch

    const dim3 blk(256);
    const int ewGrid = (B * H + 255) / 256;                  // 64

    init_state<<<ewGrid, blk, 0, stream>>>(h0f, h1f, h0b, h1b);

    if (fast) {
        // One-time weight conversion (amortized over the whole graph).
        auto cv = [&](const float* s, bf16* d, int n) {
            cvt_w<<<(n + 255) / 256, blk, 0, stream>>>(s, d, n);
        };
        cv(wxz0, wz0b, H * E); cv(wxr0, wr0b, H * E); cv(wxn0, wn0b, H * E);
        cv(whz0, uz0b, H * H); cv(whr0, ur0b, H * H); cv(whn0, un0b, H * H);
        cv(wxz1, wz1b, H * H); cv(wxr1, wr1b, H * H); cv(wxn1, wn1b, H * H);
        cv(whz1, uz1b, H * H); cv(whr1, ur1b, H * H); cv(whn1, un1b, H * H);
        cv(fcw,  fcwb, V * H);

        // Recurrence-independent layer-0 x-part for ALL timesteps:
        // X0[4096, 3072] = xall @ [Wxz0;Wxr0;Wxn0]^T + bias  (whole-GPU GEMM)
        xgather_all<<<(B * S * E + 255) / 256, blk, 0, stream>>>(x, emb, xall);
        gemm_gates<bf16, 4><<<256 * (3072 / 64) / 8, blk, 0, stream>>>(
            xall, E, (const bf16*)nullptr, 0,
            wz0b, wr0b, wn0b, nullptr, nullptr, nullptr,
            bxz0, bxr0, bxn0, X0, 3 * H);

        for (int t = 0; t < S; ++t) {
            const float* x0s = X0 + (size_t)t * B * 3 * H;
            // layer 0: h-parts only (K=1024)
            gemm_gates<bf16, 4><<<4, blk, 0, stream>>>(
                h0b, H, (const bf16*)nullptr, 0,
                uz0b, ur0b, nullptr, nullptr, nullptr, nullptr,
                nullptr, nullptr, nullptr, gzr0, 2 * H);
            gru_zr_ew<<<ewGrid, blk, 0, stream>>>(gzr0, x0s, h0f, z0, rh0);
            gemm_gates<bf16, 4><<<2, blk, 0, stream>>>(
                rh0, H, (const bf16*)nullptr, 0,
                un0b, nullptr, nullptr, nullptr, nullptr, nullptr,
                nullptr, nullptr, nullptr, gn0, H);
            gru_comb<<<ewGrid, blk, 0, stream>>>(gn0, x0s, z0, h0f, h0b, (bf16*)nullptr);
            // layer 1: x-part (h0) and h-part fused, K=2048
            gemm_gates<bf16, 4><<<4, blk, 0, stream>>>(
                h0b, H, h1b, H,
                wz1b, wr1b, nullptr, uz1b, ur1b, nullptr,
                bxz1, bxr1, nullptr, gzr1, 2 * H);
            gru_zr_ew<<<ewGrid, blk, 0, stream>>>(gzr1, (const float*)nullptr, h1f, z1, rh1);
            gemm_gates<bf16, 4><<<2, blk, 0, stream>>>(
                h0b, H, rh1, H,
                wn1b, nullptr, nullptr, un1b, nullptr, nullptr,
                bxn1, nullptr, nullptr, gn1, H);
            gru_comb<<<ewGrid, blk, 0, stream>>>(gn1, (const float*)nullptr, z1, h1f, h1b,
                                                 outs + (size_t)t * B * H);
        }
        fc_gemm<bf16><<<8000, blk, 0, stream>>>(outs, fcwb, fcb, out);
    } else {
        // Fallback: inline f32 weight conversion, per-step x gather.
        for (int t = 0; t < S; ++t) {
            xgather<<<(B * E + 255) / 256, blk, 0, stream>>>(x, emb, t, xbf);
            gemm_gates<float, 4><<<4, blk, 0, stream>>>(
                xbf, E, h0b, H,
                wxz0, wxr0, nullptr, whz0, whr0, nullptr,
                bxz0, bxr0, nullptr, gzr0, 2 * H);
            gru_zr_ew<<<ewGrid, blk, 0, stream>>>(gzr0, (const float*)nullptr, h0f, z0, rh0);
            gemm_gates<float, 4><<<2, blk, 0, stream>>>(
                xbf, E, rh0, H,
                wxn0, nullptr, nullptr, whn0, nullptr, nullptr,
                bxn0, nullptr, nullptr, gn0, H);
            gru_comb<<<ewGrid, blk, 0, stream>>>(gn0, (const float*)nullptr, z0, h0f, h0b, (bf16*)nullptr);
            gemm_gates<float, 4><<<4, blk, 0, stream>>>(
                h0b, H, h1b, H,
                wxz1, wxr1, nullptr, whz1, whr1, nullptr,
                bxz1, bxr1, nullptr, gzr1, 2 * H);
            gru_zr_ew<<<ewGrid, blk, 0, stream>>>(gzr1, (const float*)nullptr, h1f, z1, rh1);
            gemm_gates<float, 4><<<2, blk, 0, stream>>>(
                h0b, H, rh1, H,
                wxn1, nullptr, nullptr, whn1, nullptr, nullptr,
                bxn1, nullptr, nullptr, gn1, H);
            gru_comb<<<ewGrid, blk, 0, stream>>>(gn1, (const float*)nullptr, z1, h1f, h1b,
                                                 outs + (size_t)t * B * H);
        }
        fc_gemm<float><<<8000, blk, 0, stream>>>(outs, fcw, fcb, out);
    }

    copy_states<<<ewGrid, blk, 0, stream>>>(h0f, h1f, out_h0, out_h1);
}
